// Decoder_55044300865889
// MI455X (gfx1250) — matrix operand order repaired and checked by reading, not compiled, not run on hardware
//
#include <hip/hip_runtime.h>
#include <hip/hip_bf16.h>

// ---------------------------------------------------------------------------
// Decoder for MI455X (gfx1250): bf16 WMMA fused dual-GEMM pipeline.
//   B=2048, H=1024, D=512, MAX_N=64, MID_DEC=768, MID_SP=512, MID_CD=512
// ---------------------------------------------------------------------------

#define Bv       2048
#define Hv       1024
#define Dv       512
#define MAXN     64
#define MID_DEC  768
#define MID_SP   512
#define MID_CD   512

// LDS layout for the main kernel (bf16, padded rows for bank-conflict-free
// ds_load_b128 fragment loads: dword stride / 4 must be odd).
#define A_STRIDE 1032            // 1024 + 8 pad  (516 dwords, 129 granules)
#define H_STRIDE 776             // 768 + 8 pad   (388 dwords,  97 granules)
#define A_ELEMS  (64 * A_STRIDE) // 66048 bf16
#define H_ELEMS  (64 * H_STRIDE) // 49664 bf16
#define SMEM_BYTES ((A_ELEMS + H_ELEMS) * 2)   // 231,424 B  (< 320 KB WGP LDS)

typedef float  v8f   __attribute__((ext_vector_type(8)));
typedef __bf16 v16bf __attribute__((ext_vector_type(16)));

union FragU {
    uint4  u[2];
    v16bf  v;
};

// float -> bf16 bits with round-to-nearest-even
static __device__ __forceinline__ unsigned short f2bf(float f) {
    unsigned u = __float_as_uint(f);
    unsigned r = u + 0x7FFFu + ((u >> 16) & 1u);
    return (unsigned short)(r >> 16);
}

// mish(x) = x * tanh(softplus(x));  tanh(log1p(e^x)) = t(t+2)/(t(t+2)+2), t=e^x
static __device__ __forceinline__ float mishf(float x) {
    float t = __expf(x);
    float g = t * (t + 2.0f);
    float y = g / (g + 2.0f);
    return (x > 15.0f) ? x : x * y;
}

// ---------------------------------------------------------------------------
// Pack a row-major f32 weight matrix W[K,N] into WMMA B-fragment order (bf16):
//   frag fi = ks*NT + nt ; per lane: n = nt*16 + (lane&15), kb = lane<16?0:16,
//   dword i = pack(W[ks*32+kb+2i][n], W[ks*32+kb+2i+1][n]).
// One thread per output dword.
// ---------------------------------------------------------------------------
__global__ __launch_bounds__(256)
void pack_wmma_b(const float* __restrict__ w, int N, int NT, int total_dwords,
                 unsigned* __restrict__ out) {
    int dw = blockIdx.x * 256 + threadIdx.x;
    if (dw >= total_dwords) return;
    int i    = dw & 7;
    int lane = (dw >> 3) & 31;
    int fi   = dw >> 8;
    int nt   = fi % NT;
    int ks   = fi / NT;
    int n    = nt * 16 + (lane & 15);
    int kb   = (lane < 16) ? 0 : 16;
    int k    = ks * 32 + kb + 2 * i;
    unsigned lo = f2bf(w[(size_t)k * N + n]);
    unsigned hi = f2bf(w[(size_t)(k + 1) * N + n]);
    out[dw] = lo | (hi << 16);
}

// ---------------------------------------------------------------------------
// size_pred: n[b] = clip(round( mish(LN(z@W1+b1))@W2 + b2 ), 0, 64)
// One block per batch row; each thread owns columns tid and tid+256.
// ---------------------------------------------------------------------------
__global__ __launch_bounds__(256)
void size_pred_kernel(const float* __restrict__ z, const float* __restrict__ w1,
                      const float* __restrict__ b1, const float* __restrict__ g,
                      const float* __restrict__ beta, const float* __restrict__ w2,
                      const float* __restrict__ b2, int* __restrict__ nOut) {
    __shared__ float zrow[Hv];
    __shared__ float red[256];
    const int b = blockIdx.x, tid = threadIdx.x;

    for (int k = tid; k < Hv; k += 256) zrow[k] = z[(size_t)b * Hv + k];
    __syncthreads();

    float a0 = b1[tid], a1 = b1[tid + 256];
    for (int k = 0; k < Hv; ++k) {
        float zk = zrow[k];
        a0 = fmaf(zk, w1[k * MID_SP + tid], a0);
        a1 = fmaf(zk, w1[k * MID_SP + tid + 256], a1);
    }
    // mean
    red[tid] = a0 + a1;
    __syncthreads();
    for (int s = 128; s > 0; s >>= 1) { if (tid < s) red[tid] += red[tid + s]; __syncthreads(); }
    float mean = red[0] * (1.0f / MID_SP);
    __syncthreads();
    // variance (biased)
    float d0 = a0 - mean, d1 = a1 - mean;
    red[tid] = d0 * d0 + d1 * d1;
    __syncthreads();
    for (int s = 128; s > 0; s >>= 1) { if (tid < s) red[tid] += red[tid + s]; __syncthreads(); }
    float var = red[0] * (1.0f / MID_SP);
    __syncthreads();
    float rstd = rsqrtf(var + 1e-5f);
    float h0 = mishf(d0 * rstd * g[tid]       + beta[tid]);
    float h1 = mishf(d1 * rstd * g[tid + 256] + beta[tid + 256]);
    red[tid] = h0 * w2[tid] + h1 * w2[tid + 256];
    __syncthreads();
    for (int s = 128; s > 0; s >>= 1) { if (tid < s) red[tid] += red[tid + s]; __syncthreads(); }
    if (tid == 0) {
        int n = (int)rintf(red[0] + b2[0]);
        n = n < 0 ? 0 : (n > MAXN ? MAXN : n);
        nOut[b] = n;
    }
}

// ---------------------------------------------------------------------------
// cardinality table: enc[v][:] = mish(v*cd_w1 + cd_b1) @ cd_w2 + cd_b2,
// for every possible v = 0..64 (n is an integer in [0,64]).
// ---------------------------------------------------------------------------
__global__ __launch_bounds__(256)
void enc_table_kernel(const float* __restrict__ cdw1, const float* __restrict__ cdb1,
                      const float* __restrict__ cdw2, const float* __restrict__ cdb2,
                      float* __restrict__ enc) {
    __shared__ float h[MID_CD];
    const int v = blockIdx.x, tid = threadIdx.x;
    const float nf = (float)v;
    for (int j = tid; j < MID_CD; j += 256) h[j] = mishf(nf * cdw1[j] + cdb1[j]);
    __syncthreads();
    for (int i = tid; i < Hv; i += 256) {
        float a = cdb2[i];
        for (int j = 0; j < MID_CD; ++j) a = fmaf(h[j], cdw2[(size_t)j * Hv + i], a);
        enc[(size_t)v * Hv + i] = a;
    }
}

// ---------------------------------------------------------------------------
// Main fused decoder. One workgroup (8 wave32) per batch row b.
//   A[p,k] = (z[b,k]-enc[n[b],k]) * key[p,k]       (64x1024 bf16, LDS)
//   Hm     = mish(A @ W1 + b1)                     (64x768  bf16, LDS)
//   X      = Hm @ W2 + b2, masked by p < n[b]      (64x512  f32, global)
// Both GEMMs via v_wmma_f32_16x16x32_bf16; weights pre-packed in B-fragment
// order so each B fragment is two coalesced global_load_b128 per lane.
// ---------------------------------------------------------------------------
__global__ __launch_bounds__(256)
void decoder_main(const float* __restrict__ z, const float* __restrict__ key,
                  const float* __restrict__ enc, const int* __restrict__ nArr,
                  const uint4* __restrict__ w1p, const uint4* __restrict__ w2p,
                  const float* __restrict__ db1, const float* __restrict__ db2,
                  float* __restrict__ outx, int* __restrict__ outb) {
    extern __shared__ char smem[];
    unsigned short* Alds = (unsigned short*)smem;
    unsigned short* Hlds = (unsigned short*)(smem + (size_t)A_ELEMS * 2);

    const int b   = blockIdx.x;
    const int tid = threadIdx.x;
    const int nb  = nArr[b];

    if (tid < MAXN) outb[(size_t)b * MAXN + tid] = (tid < nb) ? b : -1;

    // Stage zc row (f32) in the H region (reused later), then build A in bf16.
    float* zcf = (float*)Hlds;
    for (int k = tid; k < Hv; k += 256)
        zcf[k] = z[(size_t)b * Hv + k] - enc[(size_t)nb * Hv + k];
    __syncthreads();
    for (int idx = tid; idx < MAXN * Hv; idx += 256) {
        int p = idx >> 10, k = idx & (Hv - 1);
        Alds[p * A_STRIDE + k] = f2bf(zcf[k] * key[p * Hv + k]);
    }
    __syncthreads();

    const int lane  = tid & 31;
    const int wave  = tid >> 5;
    const int mt    = wave & 3;        // M tile 0..3 (rows p = mt*16..mt*16+15)
    const int half  = wave >> 2;       // N half
    const int mrowA = mt * 16 + (lane & 15);
    const int kbA   = (lane < 16) ? 0 : 16;  // A-fragment K sub-offset
    const int hi8   = (lane < 16) ? 0 : 8;   // C/D row offset for upper lanes
    const int ncl   = lane & 15;
    const v8f  vzero = (v8f){0.f, 0.f, 0.f, 0.f, 0.f, 0.f, 0.f, 0.f};

    // ---------------- GEMM1: [64,1024] x [1024,768] -> mish -> Hlds --------
    for (int ch = 0; ch < 3; ++ch) {
        const int nbase = half * 24 + ch * 8;
        v8f acc[8];
#pragma unroll
        for (int t = 0; t < 8; ++t) acc[t] = vzero;

        for (int ks = 0; ks < 32; ++ks) {
            FragU a;
            const unsigned short* ap = Alds + mrowA * A_STRIDE + ks * 32 + kbA;
            a.u[0] = *(const uint4*)ap;
            a.u[1] = *(const uint4*)(ap + 8);
#pragma unroll
            for (int t = 0; t < 8; ++t) {
                const int fi = ks * 48 + nbase + t;
                FragU bb;
                bb.u[0] = w1p[fi * 64 + lane * 2];
                bb.u[1] = w1p[fi * 64 + lane * 2 + 1];
                acc[t] = __builtin_amdgcn_wmma_f32_16x16x32_bf16(
                    false, a.v, false, bb.v, (short)0, acc[t], false, false);
            }
        }
#pragma unroll
        for (int t = 0; t < 8; ++t) {
            const int   ncol = (nbase + t) * 16 + ncl;
            const float bias = db1[ncol];
#pragma unroll
            for (int j = 0; j < 8; ++j) {
                const int row = mt * 16 + j + hi8;
                Hlds[row * H_STRIDE + ncol] = f2bf(mishf(acc[t][j] + bias));
            }
        }
    }
    __syncthreads();

    // ---------------- GEMM2: [64,768] x [768,512] + bias, mask, store ------
    for (int ch = 0; ch < 2; ++ch) {
        const int nbase = half * 16 + ch * 8;
        v8f acc[8];
#pragma unroll
        for (int t = 0; t < 8; ++t) acc[t] = vzero;

        for (int ks = 0; ks < 24; ++ks) {
            FragU a;
            const unsigned short* hp = Hlds + mrowA * H_STRIDE + ks * 32 + kbA;
            a.u[0] = *(const uint4*)hp;
            a.u[1] = *(const uint4*)(hp + 8);
#pragma unroll
            for (int t = 0; t < 8; ++t) {
                const int fi = ks * 32 + nbase + t;
                FragU bb;
                bb.u[0] = w2p[fi * 64 + lane * 2];
                bb.u[1] = w2p[fi * 64 + lane * 2 + 1];
                acc[t] = __builtin_amdgcn_wmma_f32_16x16x32_bf16(
                    false, a.v, false, bb.v, (short)0, acc[t], false, false);
            }
        }
#pragma unroll
        for (int t = 0; t < 8; ++t) {
            const int   ncol = (nbase + t) * 16 + ncl;
            const float bias = db2[ncol];
#pragma unroll
            for (int j = 0; j < 8; ++j) {
                const int   p = mt * 16 + j + hi8;
                const float v = (p < nb) ? (acc[t][j] + bias) : 0.0f;
                outx[((size_t)b * MAXN + p) * Dv + ncol] = v;
            }
        }
    }
}

// ---------------------------------------------------------------------------
// Host launcher
// ---------------------------------------------------------------------------
extern "C" void kernel_launch(void* const* d_in, const int* in_sizes, int n_in,
                              void* d_out, int out_size, void* d_ws, size_t ws_size,
                              hipStream_t stream) {
    (void)in_sizes; (void)n_in; (void)out_size; (void)ws_size;

    const float* z        = (const float*)d_in[0];
    const float* key      = (const float*)d_in[1];
    const float* sp_w1    = (const float*)d_in[2];
    const float* sp_b1    = (const float*)d_in[3];
    const float* sp_g     = (const float*)d_in[4];
    const float* sp_beta  = (const float*)d_in[5];
    const float* sp_w2    = (const float*)d_in[6];
    const float* sp_b2    = (const float*)d_in[7];
    const float* cd_w1    = (const float*)d_in[8];
    const float* cd_b1    = (const float*)d_in[9];
    const float* cd_w2    = (const float*)d_in[10];
    const float* cd_b2    = (const float*)d_in[11];
    const float* dec_w1   = (const float*)d_in[12];
    const float* dec_b1   = (const float*)d_in[13];
    const float* dec_w2   = (const float*)d_in[14];
    const float* dec_b2   = (const float*)d_in[15];

    // Workspace carve-up (all 16B aligned)
    char* ws = (char*)d_ws;
    const size_t W1P_BYTES = (size_t)32 * 48 * 32 * 32;  // 1,572,864
    const size_t W2P_BYTES = (size_t)24 * 32 * 32 * 32;  //   786,432
    const size_t ENC_BYTES = (size_t)(MAXN + 1) * Hv * 4;//   266,240
    unsigned* W1p = (unsigned*)(ws);
    unsigned* W2p = (unsigned*)(ws + W1P_BYTES);
    float*    enc = (float*)(ws + W1P_BYTES + W2P_BYTES);
    int*      nArr= (int*)(ws + W1P_BYTES + W2P_BYTES + ENC_BYTES);

    float* outx = (float*)d_out;
    int*   outb = (int*)(outx + (size_t)Bv * MAXN * Dv);

    // Allow >64KB dynamic LDS for the fused kernel.
    (void)hipFuncSetAttribute((const void*)decoder_main,
                              hipFuncAttributeMaxDynamicSharedMemorySize,
                              SMEM_BYTES);

    // 1) pack weights into WMMA B-fragment layout (bf16)
    pack_wmma_b<<<(32 * 48 * 256) / 256, 256, 0, stream>>>(dec_w1, MID_DEC, 48,
                                                           32 * 48 * 256, W1p);
    pack_wmma_b<<<(24 * 32 * 256) / 256, 256, 0, stream>>>(dec_w2, Dv, 32,
                                                           24 * 32 * 256, W2p);
    // 2) n[b] via size_pred MLP
    size_pred_kernel<<<Bv, 256, 0, stream>>>(z, sp_w1, sp_b1, sp_g, sp_beta,
                                             sp_w2, sp_b2, nArr);
    // 3) cardinality-encoding table for all 65 possible n values
    enc_table_kernel<<<MAXN + 1, 256, 0, stream>>>(cd_w1, cd_b1, cd_w2, cd_b2, enc);
    // 4) fused bf16-WMMA decoder
    decoder_main<<<Bv, 256, SMEM_BYTES, stream>>>(z, key, enc, nArr, (const uint4*)W1p,
                                                  (const uint4*)W2p, dec_b1, dec_b2,
                                                  outx, outb);
}
